// myGCNmodule_49048526520465
// MI455X (gfx1250) — matrix operand order repaired and checked by reading, not compiled, not run on hardware
//
#include <hip/hip_runtime.h>
#include <hip/hip_bf16.h>

#define E_DIM 160
#define NSEQ  4096
#define PITCH 176            // bf16 elements, keeps rows 16B-aligned, dodges bank conflicts
#define EE    (E_DIM * E_DIM)

#if __has_builtin(__builtin_amdgcn_global_load_async_to_lds_b128) && \
    __has_builtin(__builtin_amdgcn_s_wait_asynccnt)
#define HAVE_ASYNC_LDS 1
#endif

typedef __bf16 v16bf __attribute__((ext_vector_type(16)));
typedef __bf16 v8bf  __attribute__((ext_vector_type(8)));
typedef float  v8f   __attribute__((ext_vector_type(8)));
typedef int    v4i   __attribute__((vector_size(4 * sizeof(int))));

__device__ __forceinline__ v8f zero8() {
  v8f z = {0.f, 0.f, 0.f, 0.f, 0.f, 0.f, 0.f, 0.f};
  return z;
}

__device__ __forceinline__ v8f wmma_bf16(v16bf a, v16bf b, v8f c) {
  // D = A(16x32) * B(32x16) + C, f32 accum
  return __builtin_amdgcn_wmma_f32_16x16x32_bf16(false, a, false, b, (short)0, c, false, false);
}

// A fragment: 16x32 bf16 tile from row-major [row0.., k0..k0+31].
// ISA layout: lane l holds row (l&15); elems 0..7 = K (l>>4)*8 + 0..7, elems 8..15 = +16.
__device__ __forceinline__ v16bf loadA(const __bf16* base, int pitch, int row0, int k0) {
  const int lane = threadIdx.x & 31;
  const __bf16* p = base + (row0 + (lane & 15)) * pitch + k0 + ((lane >> 4) << 3);
  v16bf f;
  uint4* fp = reinterpret_cast<uint4*>(&f);
  fp[0] = *reinterpret_cast<const uint4*>(p);
  fp[1] = *reinterpret_cast<const uint4*>(p + 16);
  return f;
}

// B fragment: 32x16 with B[k][n] = W[n0+n][k0+k], W row-major (n,k).
// ISA layout: lane l holds column (l&15); elems 0..7 = K (l>>4)*8 + 0..7, elems 8..15 = +16.
__device__ __forceinline__ v16bf loadB(const __bf16* base, int pitch, int n0, int k0) {
  const int lane = threadIdx.x & 31;
  const __bf16* p = base + (size_t)(n0 + (lane & 15)) * pitch + k0 + ((lane >> 4) << 3);
  v16bf f;
  uint4* fp = reinterpret_cast<uint4*>(&f);
  fp[0] = *reinterpret_cast<const uint4*>(p);
  fp[1] = *reinterpret_cast<const uint4*>(p + 16);
  return f;
}

// f32 C/D tile: lane l holds (M = (l>>4)*8 + r, N = l&15)
__device__ __forceinline__ void storeRM(__bf16* base, int pitch, int row0, int col0, v8f acc) {
  const int lane = threadIdx.x & 31;
  __bf16* p = base + (row0 + ((lane >> 4) << 3)) * pitch + col0 + (lane & 15);
#pragma unroll
  for (int r = 0; r < 8; ++r) p[r * pitch] = (__bf16)acc[r];
}

// transposed store: element (M,N) -> base[N][M]; per-lane 8 contiguous -> one b128 store
__device__ __forceinline__ void storeTR(__bf16* base, int pitch, int row0, int col0, v8f acc) {
  const int lane = threadIdx.x & 31;
  v8bf t;
#pragma unroll
  for (int r = 0; r < 8; ++r) t[r] = (__bf16)acc[r];
  *reinterpret_cast<v8bf*>(base + (col0 + (lane & 15)) * pitch + row0 + ((lane >> 4) << 3)) = t;
}

__device__ __forceinline__ void storeF32(float* base, int pitch, int row0, int col0, v8f acc) {
  const int lane = threadIdx.x & 31;
  float* p = base + (row0 + ((lane >> 4) << 3)) * pitch + col0 + (lane & 15);
#pragma unroll
  for (int r = 0; r < 8; ++r) p[r * pitch] = acc[r];
}

template <int NCOLS>
__device__ __forceinline__ void softmax_rows(const float* sS, __bf16* sP, int nrows, float scale) {
  const int tid = threadIdx.x;
  for (int r = tid; r < nrows; r += 256) {
    const float* srow = sS + r * 32;
    float v[NCOLS];
    float mx = -3.4e38f;
#pragma unroll
    for (int j = 0; j < NCOLS; ++j) { v[j] = srow[j] * scale; mx = fmaxf(mx, v[j]); }
    float sum = 0.f;
#pragma unroll
    for (int j = 0; j < NCOLS; ++j) { v[j] = __expf(v[j] - mx); sum += v[j]; }
    const float inv = 1.f / sum;
    __bf16* prow = sP + r * 32;
#pragma unroll
    for (int j = 0; j < NCOLS; ++j) prow[j] = (__bf16)(v[j] * inv);
#pragma unroll
    for (int j = NCOLS; j < 32; ++j) prow[j] = (__bf16)0.f;  // zero-pad K for P@V
  }
}

// Gather 16 indexed rows of X (f32, 640B each) into bf16 sSeq.
// Preferred path: GLOBAL_LOAD_ASYNC_TO_LDS_B128 into an f32 staging buffer (ASYNCcnt),
// then LDS->LDS bf16 conversion. Fallback: synchronous b128 loads + convert.
__device__ __forceinline__ void gather_seq(const float* __restrict__ X, const int* __restrict__ idxrow,
                                           int b, int n, int* sIdx, float* sStage, __bf16* sSeq) {
  const int tid = threadIdx.x;
  if (tid < 15) sIdx[tid + 1] = idxrow[tid];
  if (tid == 15) sIdx[0] = n;
  __syncthreads();
#ifdef HAVE_ASYNC_LDS
  for (int e = tid; e < 16 * 40; e += 256) {
    const int r = e / 40, c4 = (e - r * 40) * 4;
    const float* g = X + ((size_t)b * NSEQ + sIdx[r]) * E_DIM + c4;
    float* l = sStage + r * E_DIM + c4;
    __builtin_amdgcn_global_load_async_to_lds_b128(
        (__attribute__((address_space(1))) v4i*)g,
        (__attribute__((address_space(3))) v4i*)l, 0, 0);
  }
  __builtin_amdgcn_s_wait_asynccnt(0);
  __syncthreads();
  for (int e = tid; e < 16 * 40; e += 256) {
    const int r = e / 40, c4 = (e - r * 40) * 4;
    const float4 v = *reinterpret_cast<const float4*>(sStage + r * E_DIM + c4);
    __bf16* d = sSeq + r * PITCH + c4;
    d[0] = (__bf16)v.x; d[1] = (__bf16)v.y; d[2] = (__bf16)v.z; d[3] = (__bf16)v.w;
  }
#else
  (void)sStage;
  for (int e = tid; e < 16 * 40; e += 256) {
    const int r = e / 40, c4 = (e - r * 40) * 4;
    const float4 v = *reinterpret_cast<const float4*>(X + ((size_t)b * NSEQ + sIdx[r]) * E_DIM + c4);
    __bf16* d = sSeq + r * PITCH + c4;
    d[0] = (__bf16)v.x; d[1] = (__bf16)v.y; d[2] = (__bf16)v.z; d[3] = (__bf16)v.w;
  }
#endif
  __syncthreads();
}

// One 5-head attention layer over a 16-row sequence already staged in sSeq (bf16).
__device__ __forceinline__ void attn_layer16(
    const __bf16* __restrict__ WQ, const __bf16* __restrict__ WK, const __bf16* __restrict__ WV,
    const float* __restrict__ in_b, const __bf16* __restrict__ WO, const float* __restrict__ out_b,
    int destRow, const __bf16* sSeq, __bf16* sQ, __bf16* sK, __bf16* sVT,
    __bf16* sP, float* sS, __bf16* sO, __bf16* sA32) {
  const int tid = threadIdx.x;
  const int wave = tid >> 5;
  const int lane15 = tid & 15;

  // fused QKV projections: 3 mats x 10 n-tiles, K=160
  for (int job = wave; job < 30; job += 8) {
    const int mat = job / 10, nt = job % 10;
    const __bf16* W = (mat == 0) ? WQ : (mat == 1) ? WK : WV;
    v8f acc = zero8();
#pragma unroll
    for (int kk = 0; kk < 5; ++kk)
      acc = wmma_bf16(loadA(sSeq, PITCH, 0, kk * 32), loadB(W, E_DIM, nt * 16, kk * 32), acc);
    const float bb = in_b[mat * E_DIM + nt * 16 + lane15];
#pragma unroll
    for (int r = 0; r < 8; ++r) acc[r] += bb;
    if (mat == 0)      storeRM(sQ, PITCH, 0, nt * 16, acc);
    else if (mat == 1) storeRM(sK, PITCH, 0, nt * 16, acc);
    else               storeTR(sVT, 32, 0, nt * 16, acc);   // V stored transposed
  }
  __syncthreads();

  // scores: per head S_h = qh_h (16x32) @ kh_h^T  -> one WMMA each
  for (int h = wave; h < 5; h += 8) {
    v8f acc = zero8();
    acc = wmma_bf16(loadA(sQ, PITCH, 0, 32 * h), loadB(sK, PITCH, 0, 32 * h), acc);
    storeF32(sS, 32, h * 16, 0, acc);
  }
  __syncthreads();

  softmax_rows<16>(sS, sP, 80, 0.17677669529663687f);  // 1/sqrt(32)
  __syncthreads();

  // O_h = P_h (16x32, zero-padded) @ vh_h : 5 heads x 2 n-tiles
  for (int job = wave; job < 10; job += 8) {
    const int h = job >> 1, t = job & 1;
    v8f acc = zero8();
    acc = wmma_bf16(loadA(sP, 32, h * 16, 0), loadB(sVT, 32, h * 32 + t * 16, 0), acc);
    storeRM(sO, PITCH, 0, h * 32 + t * 16, acc);
  }
  __syncthreads();

  // output projection -> sA32 rows [destRow, destRow+16)
  for (int nt = wave; nt < 10; nt += 8) {
    v8f acc = zero8();
#pragma unroll
    for (int kk = 0; kk < 5; ++kk)
      acc = wmma_bf16(loadA(sO, PITCH, 0, kk * 32), loadB(WO, E_DIM, nt * 16, kk * 32), acc);
    const float bb = out_b[nt * 16 + lane15];
#pragma unroll
    for (int r = 0; r < 8; ++r) acc[r] += bb;
    storeRM(sA32, PITCH, destRow, nt * 16, acc);
  }
  __syncthreads();
}

__global__ __launch_bounds__(256) void node_attn_kernel(
    const float* __restrict__ X, const int* __restrict__ in_idx, const int* __restrict__ out_idx,
    const float* __restrict__ i_in_b, const float* __restrict__ i_out_b,
    const float* __restrict__ o_in_b, const float* __restrict__ o_out_b,
    const float* __restrict__ f_in_b, const float* __restrict__ f_out_b,
    const float* __restrict__ lin_b,
    const __bf16* __restrict__ WQi, const __bf16* __restrict__ WKi, const __bf16* __restrict__ WVi,
    const __bf16* __restrict__ WQo, const __bf16* __restrict__ WKo, const __bf16* __restrict__ WVo,
    const __bf16* __restrict__ WQf, const __bf16* __restrict__ WKf, const __bf16* __restrict__ WVf,
    const __bf16* __restrict__ WOi, const __bf16* __restrict__ WOo, const __bf16* __restrict__ WOf,
    const __bf16* __restrict__ WL, float* __restrict__ out) {
  __shared__ __align__(16) int    sIdx[16];
  __shared__ __align__(16) __bf16 sSeq[16 * PITCH];
  __shared__ __align__(16) __bf16 sQ[32 * PITCH];
  __shared__ __align__(16) __bf16 sK[32 * PITCH];
  __shared__ __align__(16) __bf16 sO[32 * PITCH];
  __shared__ __align__(16) __bf16 sA32[32 * PITCH];
  __shared__ __align__(16) __bf16 sVT[160 * 32];   // V^T, K padded to 32
  __shared__ __align__(16) __bf16 sP[80 * 32];     // probs, zero-padded K
  __shared__ __align__(16) float  sS[80 * 32];     // f32 scores
  __shared__ __align__(16) float  sF[32 * 160];    // f-layer output; doubles as async gather stage
  __shared__ __align__(16) float  sPooled[160];

  const int node = blockIdx.x;
  const int b = node >> 12;       // N = 4096
  const int n = node & 4095;
  const int tid = threadIdx.x;
  const int wave = tid >> 5;
  const int lane15 = tid & 15;

  // warm L2 with weights (global_prefetch_b8)
  {
    const char* wp = reinterpret_cast<const char*>(WQi);
    for (size_t off = (size_t)tid * 128; off < (size_t)13 * EE * 2; off += 256 * 128)
      __builtin_prefetch(wp + off, 0, 1);
  }
  // zero V^T once (avoid NaN garbage in padded K columns)
  for (int i = tid; i < (160 * 32 * 2) / 16; i += 256)
    reinterpret_cast<uint4*>(sVT)[i] = make_uint4(0, 0, 0, 0);

  // ---- in-neighborhood attention layer -> sA32 rows 0..15
  gather_seq(X, in_idx + (size_t)node * 15, b, n, sIdx, sF, sSeq);
  attn_layer16(WQi, WKi, WVi, i_in_b, WOi, i_out_b, 0, sSeq, sQ, sK, sVT, sP, sS, sO, sA32);

  // ---- out-neighborhood attention layer -> sA32 rows 16..31
  gather_seq(X, out_idx + (size_t)node * 15, b, n, sIdx, sF, sSeq);
  attn_layer16(WQo, WKo, WVo, o_in_b, WOo, o_out_b, 16, sSeq, sQ, sK, sVT, sP, sS, sO, sA32);

  // ---- final single-head attention over 32x160
  for (int job = wave; job < 60; job += 8) {  // 3 mats x 2 m-tiles x 10 n-tiles
    const int mat = job / 20, rem = job % 20, mt = rem / 10, nt = rem % 10;
    const __bf16* W = (mat == 0) ? WQf : (mat == 1) ? WKf : WVf;
    v8f acc = zero8();
#pragma unroll
    for (int kk = 0; kk < 5; ++kk)
      acc = wmma_bf16(loadA(sA32, PITCH, mt * 16, kk * 32), loadB(W, E_DIM, nt * 16, kk * 32), acc);
    const float bb = f_in_b[mat * E_DIM + nt * 16 + lane15];
#pragma unroll
    for (int r = 0; r < 8; ++r) acc[r] += bb;
    if (mat == 0)      storeRM(sQ, PITCH, mt * 16, nt * 16, acc);
    else if (mat == 1) storeRM(sK, PITCH, mt * 16, nt * 16, acc);
    else               storeTR(sVT, 32, mt * 16, nt * 16, acc);
  }
  __syncthreads();

  for (int job = wave; job < 4; job += 8) {   // S (32x32), K=160
    const int mt = job >> 1, ntt = job & 1;
    v8f acc = zero8();
#pragma unroll
    for (int kk = 0; kk < 5; ++kk)
      acc = wmma_bf16(loadA(sQ, PITCH, mt * 16, kk * 32), loadB(sK, PITCH, ntt * 16, kk * 32), acc);
    storeF32(sS, 32, mt * 16, ntt * 16, acc);
  }
  __syncthreads();

  softmax_rows<32>(sS, sP, 32, 0.07905694150420949f);  // 1/sqrt(160)
  __syncthreads();

  for (int job = wave; job < 20; job += 8) {  // O = P(32x32) @ V(32x160)
    const int mt = job / 10, nt = job % 10;
    v8f acc = zero8();
    acc = wmma_bf16(loadA(sP, 32, mt * 16, 0), loadB(sVT, 32, nt * 16, 0), acc);
    storeRM(sO, PITCH, mt * 16, nt * 16, acc);
  }
  __syncthreads();

  for (int job = wave; job < 20; job += 8) {  // F = O @ f_out_w^T + b (f32)
    const int mt = job / 10, nt = job % 10;
    v8f acc = zero8();
#pragma unroll
    for (int kk = 0; kk < 5; ++kk)
      acc = wmma_bf16(loadA(sO, PITCH, mt * 16, kk * 32), loadB(WOf, E_DIM, nt * 16, kk * 32), acc);
    const float bb = f_out_b[nt * 16 + lane15];
#pragma unroll
    for (int r = 0; r < 8; ++r) acc[r] += bb;
    storeF32(sF, 160, mt * 16, nt * 16, acc);
  }
  __syncthreads();

  // column max-pool over the 32 rows
  if (tid < 160) {
    float m = sF[tid];
#pragma unroll 4
    for (int r = 1; r < 32; ++r) m = fmaxf(m, sF[r * 160 + tid]);
    sPooled[tid] = m;
  }
  __syncthreads();

  // final linear + ELU, f32 output
  if (tid < 160) {
    const __bf16* wrow = WL + (size_t)tid * E_DIM;
    float s = lin_b[tid];
#pragma unroll 4
    for (int j = 0; j < 160; ++j) s += sPooled[j] * (float)wrow[j];
    out[(size_t)node * E_DIM + tid] = (s > 0.f) ? s : (__expf(s) - 1.f);
  }
}

// ---- setup: fuse (pre-attn projection @ qkv in-proj) into one bf16 weight per matrix
__global__ void combine_weights_kernel(const float* __restrict__ in_w, const float* __restrict__ wq,
                                       const float* __restrict__ wk, const float* __restrict__ wv,
                                       __bf16* __restrict__ dst) {
  const int idx = blockIdx.x * 256 + threadIdx.x;
  if (idx >= 3 * EE) return;
  const int mat = idx / EE, r = idx - mat * EE, n = r / E_DIM, k = r - n * E_DIM;
  const float* a = in_w + (size_t)(mat * E_DIM + n) * E_DIM;
  const float* w = ((mat == 0) ? wq : (mat == 1) ? wk : wv) + (size_t)k * E_DIM;
  float s = 0.f;
#pragma unroll 4
  for (int j = 0; j < E_DIM; ++j) s += a[j] * w[j];
  dst[idx] = (__bf16)s;
}

__global__ void convert_bf16_kernel(const float* __restrict__ src, __bf16* __restrict__ dst, int nelem) {
  const int i = blockIdx.x * 256 + threadIdx.x;
  if (i < nelem) dst[i] = (__bf16)src[i];
}

extern "C" void kernel_launch(void* const* d_in, const int* in_sizes, int n_in,
                              void* d_out, int out_size, void* d_ws, size_t ws_size,
                              hipStream_t stream) {
  (void)in_sizes; (void)n_in; (void)out_size; (void)ws_size;
  const float* X       = (const float*)d_in[0];
  const int*   in_idx  = (const int*)d_in[1];
  const int*   out_idx = (const int*)d_in[2];
  const float* iWq = (const float*)d_in[3];
  const float* iWk = (const float*)d_in[4];
  const float* iWv = (const float*)d_in[5];
  const float* i_in_w  = (const float*)d_in[6];
  const float* i_in_b  = (const float*)d_in[7];
  const float* i_out_w = (const float*)d_in[8];
  const float* i_out_b = (const float*)d_in[9];
  const float* oWq = (const float*)d_in[10];
  const float* oWk = (const float*)d_in[11];
  const float* oWv = (const float*)d_in[12];
  const float* o_in_w  = (const float*)d_in[13];
  const float* o_in_b  = (const float*)d_in[14];
  const float* o_out_w = (const float*)d_in[15];
  const float* o_out_b = (const float*)d_in[16];
  const float* f_in_w  = (const float*)d_in[17];
  const float* f_in_b  = (const float*)d_in[18];
  const float* f_out_w = (const float*)d_in[19];
  const float* f_out_b = (const float*)d_in[20];
  const float* lin_w   = (const float*)d_in[21];
  const float* lin_b   = (const float*)d_in[22];

  __bf16* wsb = (__bf16*)d_ws;   // 13 * E*E bf16 = 650 KB
  __bf16* WQi = wsb + 0 * EE;
  __bf16* WKi = wsb + 1 * EE;
  __bf16* WVi = wsb + 2 * EE;
  __bf16* WQo = wsb + 3 * EE;
  __bf16* WKo = wsb + 4 * EE;
  __bf16* WVo = wsb + 5 * EE;
  __bf16* WQf = wsb + 6 * EE;
  __bf16* WKf = wsb + 7 * EE;
  __bf16* WVf = wsb + 8 * EE;
  __bf16* WOi = wsb + 9 * EE;
  __bf16* WOo = wsb + 10 * EE;
  __bf16* WOf = wsb + 11 * EE;
  __bf16* WL  = wsb + 12 * EE;

  const int blocks3 = (3 * EE + 255) / 256;
  const int blocks1 = (EE + 255) / 256;
  combine_weights_kernel<<<blocks3, 256, 0, stream>>>(i_in_w, iWq, iWk, iWv, WQi);
  combine_weights_kernel<<<blocks3, 256, 0, stream>>>(o_in_w, oWq, oWk, oWv, WQo);
  convert_bf16_kernel<<<blocks3, 256, 0, stream>>>(f_in_w, WQf, 3 * EE);
  convert_bf16_kernel<<<blocks1, 256, 0, stream>>>(i_out_w, WOi, EE);
  convert_bf16_kernel<<<blocks1, 256, 0, stream>>>(o_out_w, WOo, EE);
  convert_bf16_kernel<<<blocks1, 256, 0, stream>>>(f_out_w, WOf, EE);
  convert_bf16_kernel<<<blocks1, 256, 0, stream>>>(lin_w, WL, EE);

  node_attn_kernel<<<8 * 4096, 256, 0, stream>>>(
      X, in_idx, out_idx, i_in_b, i_out_b, o_in_b, o_out_b, f_in_b, f_out_b, lin_b,
      WQi, WKi, WVi, WQo, WKo, WVo, WQf, WKf, WVf, WOi, WOo, WOf, WL, (float*)d_out);
}
